// GAT_NET_73332271612550
// MI455X (gfx1250) — matrix operand order repaired and checked by reading, not compiled, not run on hardware
//
#include <hip/hip_runtime.h>
#include <hip/hip_bf16.h>
#include <math.h>

// ---------------- problem constants ----------------
#define N_NODES   50000
#define N_EDGES   800000
#define F_IN      128
#define HID       64
#define HEADS     4
#define HD1       (HEADS * HID)   // 256
#define CLASSES   40
#define NEG_SLOPE 0.2f
#define EPS_      1e-16f

typedef __attribute__((ext_vector_type(16))) __bf16 v16bf;
typedef __attribute__((ext_vector_type(8)))  float  v8f;

// ---------------- weight conversion / transpose ----------------
// W1 [128][256] f32  ->  W1t [256][128] bf16  (column-major for B-fragment loads)
__global__ void k_convert_w1(const float* __restrict__ W1, __bf16* __restrict__ W1t) {
    int i = blockIdx.x * blockDim.x + threadIdx.x;
    if (i >= F_IN * HD1) return;
    int k = i / HD1, n = i % HD1;
    W1t[n * F_IN + k] = (__bf16)W1[i];
}

// W2 [256][40] f32 -> W2t [64][256] bf16, zero-padded cols 40..63
__global__ void k_convert_w2(const float* __restrict__ W2, __bf16* __restrict__ W2t) {
    int i = blockIdx.x * blockDim.x + threadIdx.x;
    if (i >= 64 * HD1) return;
    int n = i / HD1, k = i % HD1;
    W2t[i] = (n < CLASSES) ? (__bf16)W2[k * CLASSES + n] : (__bf16)0.0f;
}

__global__ void k_fill_f32(float* __restrict__ p, float v, int n) {
    int i = blockIdx.x * blockDim.x + threadIdx.x;
    if (i < n) p[i] = v;
}

// ---------------- WMMA GEMM with N-register-blocking ----------------
// C[M x ncols] = A[M x K](f32, cvt->bf16) * B, where Bt[n][k] is column-major bf16.
// One wave computes a 16 x (16*NBLK) strip: A fragment loaded once per K-step,
// NBLK back-to-back WMMAs. KSTEPS is compile-time so the K loop fully unrolls.
// blockDim.x = 256 (8 waves).
template <int KSTEPS, int NBLK>
__global__ void k_gemm_bf16_wmma(const float* __restrict__ A, int lda,
                                 const __bf16* __restrict__ Bt, int ldk,
                                 float* __restrict__ C, int ldc,
                                 int mtiles, int ngroups, int ncols) {
    int wave = blockIdx.x * (blockDim.x >> 5) + (threadIdx.x >> 5);
    if (wave >= mtiles * ngroups) return;          // uniform per-wave exit
    int mtile = wave / ngroups;
    int ngrp  = wave % ngroups;
    int lane  = threadIdx.x & 31;

    // A fragment (16x32 bf16): row = lane&15;
    // elems 0..7 -> K = k0 + koffA + i ; elems 8..15 -> K = k0 + koffA + 8 + i
    int arow  = lane & 15;
    int koffA = (lane >> 4) << 4;                  // 0 or 16
    // B fragment (32x16 bf16): col = lane&15; 16 consecutive K per half-wave
    int bcol  = lane & 15;
    int koffB = (lane >> 4) << 4;                  // 0 or 16

    const float* ap = A + (size_t)(mtile * 16 + arow) * lda;
    const __bf16* bp[NBLK];
#pragma unroll
    for (int j = 0; j < NBLK; ++j)
        bp[j] = Bt + (size_t)((ngrp * NBLK + j) * 16 + bcol) * ldk;

    v8f acc[NBLK];
#pragma unroll
    for (int j = 0; j < NBLK; ++j) acc[j] = (v8f){};

#pragma unroll
    for (int ks = 0; ks < KSTEPS; ++ks) {
        int k0 = ks << 5;
        v16bf a;
#pragma unroll
        for (int i = 0; i < 8; ++i) {
            a[i]     = (__bf16)ap[k0 + koffA + i];
            a[8 + i] = (__bf16)ap[k0 + koffA + 8 + i];
        }
#pragma unroll
        for (int j = 0; j < NBLK; ++j) {
            v16bf b;
#pragma unroll
            for (int i = 0; i < 16; ++i) b[i] = bp[j][k0 + koffB + i];
            acc[j] = __builtin_amdgcn_wmma_f32_16x16x32_bf16(
                false, a, false, b, (short)0, acc[j], false, false);
        }
    }

    // C/D f32 layout: VGPR r -> row (lane>>4)*8 + r, col = lane&15
    int crow0 = (lane >> 4) * 8;
#pragma unroll
    for (int j = 0; j < NBLK; ++j) {
        int col = (ngrp * NBLK + j) * 16 + bcol;
        if (col < ncols) {
#pragma unroll
            for (int r = 0; r < 8; ++r)
                C[(size_t)(mtile * 16 + crow0 + r) * ldc + col] = acc[j][r];
        }
    }
}

// ---------------- attention scalars: alpha_s/alpha_d = <h_row, a_vec> ----------------
__global__ void k_alpha(const float* __restrict__ h,
                        const float* __restrict__ a_src, const float* __restrict__ a_dst,
                        float* __restrict__ as, float* __restrict__ ad,
                        int n, int H, int D) {
    int i = blockIdx.x * blockDim.x + threadIdx.x;   // (node, head)
    if (i >= n * H) return;
    int node = i / H, hd = i % H;
    const float* hp  = h + (size_t)node * H * D + hd * D;
    const float* asp = a_src + hd * D;
    const float* adp = a_dst + hd * D;
    float s = 0.f, d = 0.f;
    for (int j = 0; j < D; ++j) { float v = hp[j]; s += v * asp[j]; d += v * adp[j]; }
    as[i] = s; ad[i] = d;
}

// ---------------- edge phase 1: e = lrelu(as[src]+ad[dst]); segment-max into m ----------------
__global__ void k_edge_e(const int* __restrict__ src, const int* __restrict__ dst,
                         const float* __restrict__ as, const float* __restrict__ ad,
                         float* __restrict__ e, float* __restrict__ m, int E, int H) {
    int i = blockIdx.x * blockDim.x + threadIdx.x;
    if (i >= E * H) return;
    int ed = i / H, hd = i % H;
    int s = src[ed], d = dst[ed];
    float v = as[s * H + hd] + ad[d * H + hd];
    v = (v > 0.f) ? v : v * NEG_SLOPE;
    e[i] = v;
    float* addr = &m[d * H + hd];                // float atomic-max via int punning
    if (v >= 0.f) atomicMax((int*)addr, __float_as_int(v));
    else          atomicMin((unsigned int*)addr, __float_as_uint(v));
}

// ---------------- edge phase 2: ex = exp(e - m[dst]); segment-sum into den ----------------
__global__ void k_edge_exp(const int* __restrict__ dst, const float* __restrict__ m,
                           float* __restrict__ e, float* __restrict__ den, int E, int H) {
    int i = blockIdx.x * blockDim.x + threadIdx.x;
    if (i >= E * H) return;
    int ed = i / H, hd = i % H;
    int d = dst[ed];
    float ex = __expf(e[i] - m[d * H + hd]);
    e[i] = ex;
    atomicAdd(&den[d * H + hd], ex);
}

// ---------------- edge phase 2.5: alpha = ex / (den[dst]+eps) (hoisted division) ----------------
__global__ void k_edge_alpha(const int* __restrict__ dst, const float* __restrict__ den,
                             float* __restrict__ e, int E, int H) {
    int i = blockIdx.x * blockDim.x + threadIdx.x;
    if (i >= E * H) return;
    int ed = i / H, hd = i % H;
    int d = dst[ed];
    e[i] = e[i] / (den[d * H + hd] + EPS_);
}

// ---------------- edge phase 3: agg[dst] += h[src] * alpha ----------------
__global__ void k_edge_msg(const int* __restrict__ src, const int* __restrict__ dst,
                           const float* __restrict__ h, const float* __restrict__ alpha,
                           float* __restrict__ agg, int E, int H, int D) {
    int F = H * D;
    int i = blockIdx.x * blockDim.x + threadIdx.x;   // (edge, feature)
    if (i >= E * F) return;
    int ed = i / F, f = i % F;
    int hd = f / D;
    int s = src[ed], d = dst[ed];
    atomicAdd(&agg[(size_t)d * F + f], h[(size_t)s * F + f] * alpha[ed * H + hd]);
}

// ---------------- bias + relu ----------------
__global__ void k_bias_relu(const float* __restrict__ in, const float* __restrict__ b,
                            float* __restrict__ out, int n, int f) {
    int i = blockIdx.x * blockDim.x + threadIdx.x;
    if (i >= n * f) return;
    float v = in[i] + b[i % f];
    out[i] = (v > 0.f) ? v : 0.f;
}

// ---------------- bias + log_softmax over CLASSES ----------------
__global__ void k_logsoftmax(const float* __restrict__ agg, const float* __restrict__ b,
                             float* __restrict__ out, int n, int C) {
    int node = blockIdx.x * blockDim.x + threadIdx.x;
    if (node >= n) return;
    const float* p = agg + (size_t)node * C;
    float mx = -INFINITY;
    for (int c = 0; c < C; ++c) { float v = p[c] + b[c]; mx = (v > mx) ? v : mx; }
    float s = 0.f;
    for (int c = 0; c < C; ++c) s += __expf(p[c] + b[c] - mx);
    float l = __logf(s) + mx;
    for (int c = 0; c < C; ++c) out[(size_t)node * C + c] = p[c] + b[c] - l;
}

// ---------------- host launcher ----------------
static inline size_t align256(size_t x) { return (x + 255) & ~(size_t)255; }

extern "C" void kernel_launch(void* const* d_in, const int* in_sizes, int n_in,
                              void* d_out, int out_size, void* d_ws, size_t ws_size,
                              hipStream_t stream) {
    (void)in_sizes; (void)n_in; (void)out_size; (void)ws_size;
    // inputs per setup_inputs() order
    const float* x      = (const float*)d_in[0];
    const int*   eidx   = (const int*)d_in[1];       // [2][E]: src row 0, dst row 1
    const float* W1     = (const float*)d_in[2];
    const float* a_src1 = (const float*)d_in[3];
    const float* a_dst1 = (const float*)d_in[4];
    const float* b1     = (const float*)d_in[5];
    const float* W2     = (const float*)d_in[6];
    const float* a_src2 = (const float*)d_in[7];
    const float* a_dst2 = (const float*)d_in[8];
    const float* b2     = (const float*)d_in[9];
    const int* src = eidx;
    const int* dst = eidx + N_EDGES;
    float* out = (float*)d_out;

    // workspace layout
    char* p = (char*)d_ws;
    size_t off = 0;
    __bf16* W1t = (__bf16*)(p + off); off = align256(off + (size_t)HD1 * F_IN * 2);
    __bf16* W2t = (__bf16*)(p + off); off = align256(off + (size_t)64 * HD1 * 2);
    float* h1   = (float*)(p + off);  off = align256(off + (size_t)N_NODES * HD1 * 4);
    float* agg1 = (float*)(p + off);  off = align256(off + (size_t)N_NODES * HD1 * 4);
    float* as1  = (float*)(p + off);  off = align256(off + (size_t)N_NODES * HEADS * 4);
    float* ad1  = (float*)(p + off);  off = align256(off + (size_t)N_NODES * HEADS * 4);
    float* m1   = (float*)(p + off);  off = align256(off + (size_t)N_NODES * HEADS * 4);
    float* den1 = (float*)(p + off);  off = align256(off + (size_t)N_NODES * HEADS * 4);
    float* e1   = (float*)(p + off);  off = align256(off + (size_t)N_EDGES * HEADS * 4);
    float* h2   = (float*)(p + off);  off = align256(off + (size_t)N_NODES * CLASSES * 4);
    float* agg2 = (float*)(p + off);  off = align256(off + (size_t)N_NODES * CLASSES * 4);
    float* as2  = (float*)(p + off);  off = align256(off + (size_t)N_NODES * 4);
    float* ad2  = (float*)(p + off);  off = align256(off + (size_t)N_NODES * 4);
    float* m2   = (float*)(p + off);  off = align256(off + (size_t)N_NODES * 4);
    float* den2 = (float*)(p + off);  off = align256(off + (size_t)N_NODES * 4);
    float* e2   = (float*)(p + off);  off = align256(off + (size_t)N_EDGES * 4);

    const int BT = 256;
    auto blocks = [](long long n, int bt) { return (unsigned)((n + bt - 1) / bt); };

    // --- init accumulators (graph-capture-safe async memsets + fill kernels) ---
    hipMemsetAsync(agg1, 0, (size_t)N_NODES * HD1 * 4, stream);
    hipMemsetAsync(agg2, 0, (size_t)N_NODES * CLASSES * 4, stream);
    hipMemsetAsync(den1, 0, (size_t)N_NODES * HEADS * 4, stream);
    hipMemsetAsync(den2, 0, (size_t)N_NODES * 4, stream);
    k_fill_f32<<<blocks(N_NODES * HEADS, BT), BT, 0, stream>>>(m1, -INFINITY, N_NODES * HEADS);
    k_fill_f32<<<blocks(N_NODES, BT), BT, 0, stream>>>(m2, -INFINITY, N_NODES);

    // --- weight conversion (bf16, column-major) ---
    k_convert_w1<<<blocks(F_IN * HD1, BT), BT, 0, stream>>>(W1, W1t);
    k_convert_w2<<<blocks(64 * HD1, BT), BT, 0, stream>>>(W2, W2t);

    // ---------------- layer 1 ----------------
    {   // 3125 m-tiles x 4 n-groups (each group = 4 tiles of 16 cols = 64 cols)
        int mtiles = N_NODES / 16, ngroups = HD1 / 64;
        long long waves = (long long)mtiles * ngroups;
        k_gemm_bf16_wmma<F_IN / 32, 4><<<blocks(waves, 8), BT, 0, stream>>>(
            x, F_IN, W1t, F_IN, h1, HD1, mtiles, ngroups, HD1);
    }
    k_alpha<<<blocks((long long)N_NODES * HEADS, BT), BT, 0, stream>>>(
        h1, a_src1, a_dst1, as1, ad1, N_NODES, HEADS, HID);
    k_edge_e<<<blocks((long long)N_EDGES * HEADS, BT), BT, 0, stream>>>(
        src, dst, as1, ad1, e1, m1, N_EDGES, HEADS);
    k_edge_exp<<<blocks((long long)N_EDGES * HEADS, BT), BT, 0, stream>>>(
        dst, m1, e1, den1, N_EDGES, HEADS);
    k_edge_alpha<<<blocks((long long)N_EDGES * HEADS, BT), BT, 0, stream>>>(
        dst, den1, e1, N_EDGES, HEADS);
    k_edge_msg<<<blocks((long long)N_EDGES * HD1, BT), BT, 0, stream>>>(
        src, dst, h1, e1, agg1, N_EDGES, HEADS, HID);
    // bias + relu, write back into h1 (h1 no longer needed) -> input of layer 2
    k_bias_relu<<<blocks((long long)N_NODES * HD1, BT), BT, 0, stream>>>(
        agg1, b1, h1, N_NODES, HD1);

    // ---------------- layer 2 ----------------
    {   // 3125 m-tiles x 1 n-group (64 padded cols)
        int mtiles = N_NODES / 16, ngroups = 1;
        long long waves = (long long)mtiles * ngroups;
        k_gemm_bf16_wmma<HD1 / 32, 4><<<blocks(waves, 8), BT, 0, stream>>>(
            h1, HD1, W2t, HD1, h2, CLASSES, mtiles, ngroups, CLASSES);
    }
    k_alpha<<<blocks(N_NODES, BT), BT, 0, stream>>>(
        h2, a_src2, a_dst2, as2, ad2, N_NODES, 1, CLASSES);
    k_edge_e<<<blocks(N_EDGES, BT), BT, 0, stream>>>(
        src, dst, as2, ad2, e2, m2, N_EDGES, 1);
    k_edge_exp<<<blocks(N_EDGES, BT), BT, 0, stream>>>(
        dst, m2, e2, den2, N_EDGES, 1);
    k_edge_alpha<<<blocks(N_EDGES, BT), BT, 0, stream>>>(
        dst, den2, e2, N_EDGES, 1);
    k_edge_msg<<<blocks((long long)N_EDGES * CLASSES, BT), BT, 0, stream>>>(
        src, dst, h2, e2, agg2, N_EDGES, 1, CLASSES);

    // bias + log_softmax -> output
    k_logsoftmax<<<blocks(N_NODES, BT), BT, 0, stream>>>(agg2, b2, out, N_NODES, CLASSES);
}
